// SubSample_79259326480739
// MI455X (gfx1250) — matrix operand order repaired and checked by reading, not compiled, not run on hardware
//
#include <hip/hip_runtime.h>
#include <hip/hip_bf16.h>

typedef __bf16 bf16_t;
typedef bf16_t bf16x16 __attribute__((ext_vector_type(16)));
typedef float  f32x8   __attribute__((ext_vector_type(8)));

// Detect CDNA5 async direct-to-LDS loads (ASYNCcnt path). Fallback: reg staging.
#ifdef __has_builtin
#if __has_builtin(__builtin_amdgcn_global_load_async_to_lds_b128) && \
    __has_builtin(__builtin_amdgcn_s_wait_asynccnt)
#define USE_ASYNC_LDS 1
#endif
#endif
#ifndef USE_ASYNC_LDS
#define USE_ASYNC_LDS 0
#endif

#if USE_ASYNC_LDS
// builtin signature: (v4i addrspace(1)* src, v4i addrspace(3)* dst, imm offset, imm cpol)
typedef int v4i_t __attribute__((vector_size(16)));
typedef __attribute__((address_space(1))) v4i_t* g_v4i;
typedef __attribute__((address_space(3))) v4i_t* l_v4i;
#define AS1_V4I(p) ((g_v4i)(p))
#define AS3_V4I(p) ((l_v4i)(p))
#endif

union Frag { bf16x16 v; uint4 q[2]; };
static_assert(sizeof(Frag) == 32, "frag size");

union U4H { uint4 q; unsigned short h[8]; };

__device__ __forceinline__ f32x8 wmma_bf16(const Frag& a, const Frag& b, f32x8 c) {
    // D = A(16x32 bf16) * B(32x16 bf16) + C(16x16 f32)
    return __builtin_amdgcn_wmma_f32_16x16x32_bf16(false, a.v, false, b.v,
                                                   (short)0, c, false, false);
}

__device__ __forceinline__ float gelu_exact(float x) {
    return 0.5f * x * (1.0f + erff(x * 0.70710678118654752440f));
}

__device__ __forceinline__ f32x8 zero8() {
    f32x8 z = {0.f, 0.f, 0.f, 0.f, 0.f, 0.f, 0.f, 0.f};
    return z;
}

// ---------------------------------------------------------------- prep kernels

__global__ void f32_to_bf16_kernel(const float* __restrict__ s, bf16_t* __restrict__ d, int n) {
    int i = blockIdx.x * blockDim.x + threadIdx.x;
    if (i < n) d[i] = (bf16_t)s[i];
}

__global__ void bn_prep_kernel(const float* __restrict__ g, const float* __restrict__ b,
                               const float* __restrict__ m, const float* __restrict__ v,
                               float2* __restrict__ out, int n) {
    int i = blockIdx.x * blockDim.x + threadIdx.x;
    if (i < n) {
        float s = g[i] * rsqrtf(v[i] + 1e-5f);
        out[i] = make_float2(s, b[i] - m[i] * s);
    }
}

// x (32,256,32,32) f32 -> xb bf16 same layout, plus xqb (32,256,16,16) = x[:,:,::2,::2]
__global__ void prep_x_kernel(const float* __restrict__ x, bf16_t* __restrict__ xb,
                              bf16_t* __restrict__ xqb) {
    int i = blockIdx.x * blockDim.x + threadIdx.x;
    if (i >= 32 * 256 * 1024) return;
    float val = x[i];
    xb[i] = (bf16_t)val;
    int hw = i & 1023;
    int h = hw >> 5, w = hw & 31;
    if (!(h & 1) && !(w & 1)) {
        int bc = i >> 10; // b*256 + c
        xqb[(bc << 8) + ((h >> 1) << 4) + (w >> 1)] = (bf16_t)val;
    }
}

// ---------------------------------------------------------------- generic WMMA GEMM + BN

enum { MODE_F32 = 0, MODE_BF16 = 1, MODE_BF16_GELU = 2, MODE_KV = 3, MODE_QT = 4 };

// out[b] = BN( A(MxK) @ Bsrc[b](KxN) )   epilogue by MODE
// 128x128 tile per workgroup, BK=32, double-buffered LDS, 8 waves each owning a
// 16-row strip x 128 cols (8 accumulators). A tile staged via async direct-to-LDS
// loads when available; B tile transposed through registers.
template <int MODE>
__global__ __launch_bounds__(256, 2) void gemm_bn_kernel(
    const bf16_t* __restrict__ A, const bf16_t* __restrict__ Bsrc, long long bStrideB,
    const float2* __restrict__ bnp, int M, int N, int K,
    void* __restrict__ out0, void* __restrict__ out1, long long bStrideO) {
    __shared__ bf16_t As[2][128][32]; // [m][k]
    __shared__ bf16_t Bs[2][128][32]; // [n][k]  (transposed during staging)

    const int tid  = threadIdx.x;
    const int wave = tid >> 5, lane = tid & 31;
    const int bIdx = blockIdx.z;
    const int m0 = blockIdx.x * 128;
    const int n0 = blockIdx.y * 128;
    const bf16_t* Bb = Bsrc + (long long)bIdx * bStrideB;

    f32x8 acc[8];
#pragma unroll
    for (int j = 0; j < 8; ++j) acc[j] = zero8();

    // A staging: thread -> row (0..127), 16-half chunk (0/16)
    const int arow  = tid >> 1;
    const int ahalf = (tid & 1) * 16;
    // B staging: thread -> k-pair kp (0..15), 8-col chunk nb8
    const int kp  = tid >> 4;        // covers k = 2kp, 2kp+1
    const int nb8 = (tid & 15) * 8;  // 8 consecutive n

#if !USE_ASYNC_LDS
    uint4 aReg0, aReg1;
#endif
    U4H bR0, bR1;

    // prologue: kick off slab 0
    {
        const bf16_t* srcA = A + (long long)(m0 + arow) * K + ahalf;
#if USE_ASYNC_LDS
        __builtin_amdgcn_global_load_async_to_lds_b128(
            AS1_V4I(srcA), AS3_V4I(&As[0][arow][ahalf]), 0, 0);
        __builtin_amdgcn_global_load_async_to_lds_b128(
            AS1_V4I(srcA + 8), AS3_V4I(&As[0][arow][ahalf + 8]), 0, 0);
#else
        aReg0 = ((const uint4*)srcA)[0];
        aReg1 = ((const uint4*)srcA)[1];
#endif
        const bf16_t* r0 = Bb + (long long)(2 * kp) * N + n0 + nb8;
        bR0.q = *(const uint4*)r0;
        bR1.q = *(const uint4*)(r0 + N);
    }

    int buf = 0;
    for (int kb = 0; kb < K; kb += 32) {
        // commit staged B registers (transposed) to LDS[buf]
#if !USE_ASYNC_LDS
        {
            uint4* dst = (uint4*)&As[buf][arow][ahalf];
            dst[0] = aReg0;
            dst[1] = aReg1;
        }
#endif
#pragma unroll
        for (int i = 0; i < 8; ++i) {
            *(unsigned int*)&Bs[buf][nb8 + i][2 * kp] =
                (unsigned int)bR0.h[i] | ((unsigned int)bR1.h[i] << 16);
        }
#if USE_ASYNC_LDS
        __builtin_amdgcn_s_wait_asynccnt(0); // A tile for this slab resident in LDS
#endif
        __syncthreads();

        // issue next slab's loads (overlaps with WMMA below)
        if (kb + 32 < K) {
            const bf16_t* srcA = A + (long long)(m0 + arow) * K + kb + 32 + ahalf;
#if USE_ASYNC_LDS
            __builtin_amdgcn_global_load_async_to_lds_b128(
                AS1_V4I(srcA), AS3_V4I(&As[buf ^ 1][arow][ahalf]), 0, 0);
            __builtin_amdgcn_global_load_async_to_lds_b128(
                AS1_V4I(srcA + 8), AS3_V4I(&As[buf ^ 1][arow][ahalf + 8]), 0, 0);
#else
            aReg0 = ((const uint4*)srcA)[0];
            aReg1 = ((const uint4*)srcA)[1];
#endif
            const bf16_t* r0 = Bb + (long long)(kb + 32 + 2 * kp) * N + n0 + nb8;
            bR0.q = *(const uint4*)r0;
            bR1.q = *(const uint4*)(r0 + N);
            if (kb + 64 < K)
                __builtin_prefetch(Bb + (long long)(kb + 64 + 2 * kp) * N + n0 + nb8, 0, 3);
        }

        // compute from LDS[buf]
        Frag af;
        {
            int row = (wave << 4) + (lane & 15);
            int kh  = (lane < 16) ? 0 : 16;
            af.q[0] = *(const uint4*)&As[buf][row][kh];
            af.q[1] = *(const uint4*)&As[buf][row][kh + 8];
        }
#pragma unroll
        for (int j = 0; j < 8; ++j) {
            Frag bf_;
            int n  = (j << 4) + (lane & 15);
            int kh = (lane < 16) ? 0 : 16;
            bf_.q[0] = *(const uint4*)&Bs[buf][n][kh];
            bf_.q[1] = *(const uint4*)&Bs[buf][n][kh + 8];
            acc[j] = wmma_bf16(af, bf_, acc[j]);
        }
        buf ^= 1;
    }

    // epilogue: C/D layout: VGPR r, lanes 0-15 -> M=r, lanes 16-31 -> M=r+8; col = lane&15
#pragma unroll
    for (int j = 0; j < 8; ++j) {
        int gn = n0 + (j << 4) + (lane & 15);
#pragma unroll
        for (int r = 0; r < 8; ++r) {
            int gm = m0 + (wave << 4) + r + ((lane >> 4) << 3);
            float2 p  = bnp[gm];
            float val = acc[j][r] * p.x + p.y;
            if (MODE == MODE_F32) {
                ((float*)out0)[(long long)bIdx * bStrideO + (long long)gm * N + gn] = val;
            } else if (MODE == MODE_BF16) {
                ((bf16_t*)out0)[(long long)bIdx * bStrideO + (long long)gm * N + gn] = (bf16_t)val;
            } else if (MODE == MODE_BF16_GELU) {
                val = gelu_exact(val);
                ((bf16_t*)out0)[(long long)bIdx * bStrideO + (long long)gm * N + gn] = (bf16_t)val;
            } else if (MODE == MODE_KV) {
                int head = gm / 160, c = gm % 160;
                if (c < 32) // k stored transposed: kt[bh][n][d]  (feeds WMMA B operand)
                    ((bf16_t*)out0)[(((long long)bIdx * 8 + head) * 1024 + gn) * 32 + c] = (bf16_t)val;
                else        // v row-major: v[bh][vi][n]           (feeds WMMA A operand)
                    ((bf16_t*)out1)[(((long long)bIdx * 8 + head) * 128 + (c - 32)) * 1024 + gn] = (bf16_t)val;
            } else { // MODE_QT : qt[bh][qi][d] transposed (feeds WMMA A operand)
                int head = gm >> 5, d = gm & 31;
                ((bf16_t*)out0)[(((long long)bIdx * 8 + head) * 256 + gn) * 32 + d] = (bf16_t)val;
            }
        }
    }
}

// ---------------------------------------------------------------- fused attention
// grid (16 q-chunks, 8 heads, 32 batch), 256 threads (8 waves)
// S = scale * q^T k   (16 x 1024, f32 in LDS) -> softmax -> att bf16 in LDS
// O = v @ att^T       (128 x 16) -> *1/rowsum -> gelu -> obuf (B,1024,256) bf16

#define ATTN_SMEM (16 * 1024 * 4 + 16 * 1024 * 2 + 16 * 32 * 2 + 16 * 4)

__global__ __launch_bounds__(256, 2) void attn_kernel(
    const bf16_t* __restrict__ qt,   // (B*8, 256, 32)
    const bf16_t* __restrict__ kt,   // (B*8, 1024, 32)
    const bf16_t* __restrict__ vb,   // (B*8, 128, 1024)
    bf16_t* __restrict__ obuf) {     // (B, 1024, 256)
    extern __shared__ char smem[];
    float*  S    = (float*)smem;                                   // [16][1024]
    bf16_t* attb = (bf16_t*)(smem + 16 * 1024 * 4);                // [16][1024]
    bf16_t* qs   = (bf16_t*)(smem + 16 * 1024 * 4 + 16 * 1024 * 2);// [16][32]
    float*  rs   = (float*)(smem + 16 * 1024 * 4 + 16 * 1024 * 2 + 16 * 32 * 2); // [16]

    const int qc   = blockIdx.x;
    const int head = blockIdx.y;
    const int b    = blockIdx.z;
    const int bh   = b * 8 + head;
    const int tid  = threadIdx.x, wave = tid >> 5, lane = tid & 31;
    const int q0   = qc * 16;

    // load q tile (16 x 32 bf16 = 1KB = 256 dwords, one per thread)
    {
        const unsigned int* src = (const unsigned int*)(qt + ((long long)bh * 256 + q0) * 32);
        ((unsigned int*)qs)[tid] = src[tid];
    }
    __syncthreads();

    // ---- phase A: S = scale * q^T k
    Frag af;
    {
        int row = lane & 15;
        int kh  = (lane < 16) ? 0 : 16;
        af.q[0] = *(const uint4*)&qs[row * 32 + kh];
        af.q[1] = *(const uint4*)&qs[row * 32 + kh + 8];
    }
    const float scale = 0.17677669529663688f; // 32^-0.5
    for (int t = wave * 8; t < wave * 8 + 8; ++t) {
        Frag bfr;
        int n = t * 16 + (lane & 15);
        const bf16_t* krow = kt + ((long long)bh * 1024 + n) * 32 + ((lane < 16) ? 0 : 16);
        bfr.q[0] = *(const uint4*)krow;
        bfr.q[1] = *(const uint4*)(krow + 8);
        f32x8 acc = zero8();
        acc = wmma_bf16(af, bfr, acc);
#pragma unroll
        for (int r = 0; r < 8; ++r) {
            int m = r + ((lane >> 4) << 3);
            S[m * 1024 + t * 16 + (lane & 15)] = acc[r] * scale;
        }
    }
    __syncthreads();

    // ---- phase B: softmax over n (1024); 16 lanes per row
    {
        int row = tid >> 4, sub = tid & 15;
        float* srow = S + row * 1024;
        float mx = -3.4e38f;
        for (int i = sub; i < 1024; i += 16) mx = fmaxf(mx, srow[i]);
#pragma unroll
        for (int d = 1; d < 16; d <<= 1) mx = fmaxf(mx, __shfl_xor(mx, d, 32));
        float sum = 0.f;
        bf16_t* arow = attb + row * 1024;
        for (int i = sub; i < 1024; i += 16) {
            float e = expf(srow[i] - mx);
            sum += e;
            arow[i] = (bf16_t)e;
        }
#pragma unroll
        for (int d = 1; d < 16; d <<= 1) sum += __shfl_xor(sum, d, 32);
        if (sub == 0) rs[row] = 1.0f / sum;
    }
    __syncthreads();

    // ---- phase C: O[vi, qi] = sum_n v[vi,n] att[qi,n]; wave = 16-row v strip
    f32x8 acc = zero8();
    const bf16_t* vrow = vb + ((long long)bh * 128 + wave * 16 + (lane & 15)) * 1024;
    const int kh_a = (lane < 16) ? 0 : 16;
    const bf16_t* arow2 = attb + (lane & 15) * 1024 + ((lane < 16) ? 0 : 16);
    for (int nb = 0; nb < 1024; nb += 32) {
        Frag a2, b2;
        a2.q[0] = *(const uint4*)(vrow + nb + kh_a);
        a2.q[1] = *(const uint4*)(vrow + nb + kh_a + 8);
        b2.q[0] = *(const uint4*)(arow2 + nb);
        b2.q[1] = *(const uint4*)(arow2 + nb + 8);
        acc = wmma_bf16(a2, b2, acc);
    }
    {
        int qiLoc = lane & 15;
        float inv = rs[qiLoc];
#pragma unroll
        for (int r = 0; r < 8; ++r) {
            int vi = wave * 16 + r + ((lane >> 4) << 3);
            float val = gelu_exact(acc[r] * inv);
            obuf[((long long)b * 1024 + head * 128 + vi) * 256 + q0 + qiLoc] = (bf16_t)val;
        }
    }
}

// ---------------------------------------------------------------- launch

static inline void* carve(char* base, size_t& off, size_t bytes) {
    void* p = base + off;
    off = (off + bytes + 255) & ~(size_t)255;
    return p;
}

extern "C" void kernel_launch(void* const* d_in, const int* in_sizes, int n_in,
                              void* d_out, int out_size, void* d_ws, size_t ws_size,
                              hipStream_t stream) {
    const float* x    = (const float*)d_in[0];
    const float* kv_w = (const float*)d_in[1];
    const float* kv_g = (const float*)d_in[2];
    const float* kv_b = (const float*)d_in[3];
    const float* kv_m = (const float*)d_in[4];
    const float* kv_v = (const float*)d_in[5];
    const float* q_w  = (const float*)d_in[6];
    const float* q_g  = (const float*)d_in[7];
    const float* q_b  = (const float*)d_in[8];
    const float* q_m  = (const float*)d_in[9];
    const float* q_v  = (const float*)d_in[10];
    const float* mg_w = (const float*)d_in[11];
    const float* mg_g = (const float*)d_in[12];
    const float* mg_b = (const float*)d_in[13];
    const float* mg_m = (const float*)d_in[14];
    const float* mg_v = (const float*)d_in[15];
    const float* f1_w = (const float*)d_in[16];
    const float* b1_g = (const float*)d_in[17];
    const float* b1_b = (const float*)d_in[18];
    const float* b1_m = (const float*)d_in[19];
    const float* b1_v = (const float*)d_in[20];
    const float* f2_w = (const float*)d_in[21];
    const float* b2_g = (const float*)d_in[22];
    const float* b2_b = (const float*)d_in[23];
    const float* b2_m = (const float*)d_in[24];
    const float* b2_v = (const float*)d_in[25];

    char* ws = (char*)d_ws;
    size_t off = 0;
    bf16_t* xb    = (bf16_t*)carve(ws, off, (size_t)32 * 256 * 1024 * 2);
    bf16_t* xqb   = (bf16_t*)carve(ws, off, (size_t)32 * 256 * 256 * 2);
    bf16_t* kvw_b = (bf16_t*)carve(ws, off, (size_t)1280 * 256 * 2);
    bf16_t* qw_b  = (bf16_t*)carve(ws, off, (size_t)256 * 256 * 2);
    bf16_t* mgw_b = (bf16_t*)carve(ws, off, (size_t)512 * 1024 * 2);
    bf16_t* f1w_b = (bf16_t*)carve(ws, off, (size_t)1024 * 512 * 2);
    bf16_t* f2w_b = (bf16_t*)carve(ws, off, (size_t)512 * 1024 * 2);
    float2* p_kv  = (float2*)carve(ws, off, (size_t)1280 * 8);
    float2* p_q   = (float2*)carve(ws, off, (size_t)256 * 8);
    float2* p_mg  = (float2*)carve(ws, off, (size_t)512 * 8);
    float2* p_b1  = (float2*)carve(ws, off, (size_t)1024 * 8);
    float2* p_b2  = (float2*)carve(ws, off, (size_t)512 * 8);
    bf16_t* kt    = (bf16_t*)carve(ws, off, (size_t)32 * 8 * 1024 * 32 * 2);
    bf16_t* vbuf  = (bf16_t*)carve(ws, off, (size_t)32 * 8 * 128 * 1024 * 2);
    bf16_t* qt    = (bf16_t*)carve(ws, off, (size_t)32 * 8 * 256 * 32 * 2);
    bf16_t* obuf  = (bf16_t*)carve(ws, off, (size_t)32 * 1024 * 256 * 2);
    bf16_t* y1    = (bf16_t*)carve(ws, off, (size_t)32 * 512 * 256 * 2);
    bf16_t* y2    = (bf16_t*)carve(ws, off, (size_t)32 * 1024 * 256 * 2);

    (void)hipFuncSetAttribute((const void*)attn_kernel,
                              hipFuncAttributeMaxDynamicSharedMemorySize, ATTN_SMEM);

    // weight conversions
    f32_to_bf16_kernel<<<(1280 * 256 + 255) / 256, 256, 0, stream>>>(kv_w, kvw_b, 1280 * 256);
    f32_to_bf16_kernel<<<(256 * 256 + 255) / 256, 256, 0, stream>>>(q_w, qw_b, 256 * 256);
    f32_to_bf16_kernel<<<(512 * 1024 + 255) / 256, 256, 0, stream>>>(mg_w, mgw_b, 512 * 1024);
    f32_to_bf16_kernel<<<(1024 * 512 + 255) / 256, 256, 0, stream>>>(f1_w, f1w_b, 1024 * 512);
    f32_to_bf16_kernel<<<(512 * 1024 + 255) / 256, 256, 0, stream>>>(f2_w, f2w_b, 512 * 1024);
    // folded BN params
    bn_prep_kernel<<<(1280 + 255) / 256, 256, 0, stream>>>(kv_g, kv_b, kv_m, kv_v, p_kv, 1280);
    bn_prep_kernel<<<1, 256, 0, stream>>>(q_g, q_b, q_m, q_v, p_q, 256);
    bn_prep_kernel<<<2, 256, 0, stream>>>(mg_g, mg_b, mg_m, mg_v, p_mg, 512);
    bn_prep_kernel<<<4, 256, 0, stream>>>(b1_g, b1_b, b1_m, b1_v, p_b1, 1024);
    bn_prep_kernel<<<2, 256, 0, stream>>>(b2_g, b2_b, b2_m, b2_v, p_b2, 512);
    // x -> bf16 (+ subsample)
    prep_x_kernel<<<(32 * 256 * 1024 + 255) / 256, 256, 0, stream>>>(x, xb, xqb);

    // kv = BN(kv_w @ x): M=1280 N=1024 K=256 -> kt (transposed) + v
    gemm_bn_kernel<MODE_KV><<<dim3(10, 8, 32), 256, 0, stream>>>(
        kvw_b, xb, (long long)256 * 1024, p_kv, 1280, 1024, 256, kt, vbuf, 0);
    // q = BN(q_w @ xq): M=256 N=256 K=256 -> qt (transposed)
    gemm_bn_kernel<MODE_QT><<<dim3(2, 2, 32), 256, 0, stream>>>(
        qw_b, xqb, (long long)256 * 256, p_q, 256, 256, 256, qt, nullptr, 0);
    // attention + gelu -> obuf (B,1024,256)
    attn_kernel<<<dim3(16, 8, 32), 256, ATTN_SMEM, stream>>>(qt, kt, vbuf, obuf);
    // mg: M=512 N=256 K=1024
    gemm_bn_kernel<MODE_BF16><<<dim3(4, 2, 32), 256, 0, stream>>>(
        mgw_b, obuf, (long long)1024 * 256, p_mg, 512, 256, 1024, y1, nullptr, (long long)512 * 256);
    // fc1 + gelu: M=1024 N=256 K=512
    gemm_bn_kernel<MODE_BF16_GELU><<<dim3(8, 2, 32), 256, 0, stream>>>(
        f1w_b, y1, (long long)512 * 256, p_b1, 1024, 256, 512, y2, nullptr, (long long)1024 * 256);
    // fc2 -> f32 output: M=512 N=256 K=1024
    gemm_bn_kernel<MODE_F32><<<dim3(4, 2, 32), 256, 0, stream>>>(
        f2w_b, y2, (long long)1024 * 256, p_b2, 512, 256, 1024, d_out, nullptr, (long long)512 * 256);
}
